// VSSBlock_8392366097072
// MI455X (gfx1250) — matrix operand order repaired and checked by reading, not compiled, not run on hardware
//
#include <hip/hip_runtime.h>
#include <hip/hip_bf16.h>
#include <stdint.h>

#define C_DIM   256
#define HWPIX   4096
#define BATCH   2
#define NPIX    (BATCH * HWPIX)     // 8192
#define DINNER  512
#define DSTATE  16
#define DTRANK  16
#define XD      (DTRANK + 2 * DSTATE)   // 48

typedef __attribute__((ext_vector_type(16))) __bf16 v16bf;
typedef __attribute__((ext_vector_type(8)))  float  v8f;

union FragU { uint4 u[2]; v16bf v; };

__device__ __forceinline__ unsigned short f2bf(float f) {
    union { float f; unsigned int u; } c; c.f = f;
    unsigned int u = c.u;
    return (unsigned short)((u + 0x7FFFu + ((u >> 16) & 1u)) >> 16);
}

// ---------------------------------------------------------------------------
// 1) x (B,C,H,W) f32 -> A0 (n=b*HW+hw, c) bf16.
//    Global -> LDS via gfx1250 async-LDS pipe (ASYNCcnt), then transposed
//    bf16 store.
// ---------------------------------------------------------------------------
__global__ __launch_bounds__(256)
void transpose_bf16_kernel(const float* __restrict__ x, unsigned short* __restrict__ A0) {
    __shared__ float tile[32][33];
    int b   = blockIdx.z;
    int hw0 = blockIdx.x * 32;
    int c0  = blockIdx.y * 32;
    int tx = threadIdx.x, ty = threadIdx.y;   // block (32,8)
#pragma unroll
    for (int cc = 0; cc < 4; ++cc) {
        int c = c0 + ty + cc * 8;
        const float* gp = &x[((size_t)b * C_DIM + c) * HWPIX + hw0 + tx];
        unsigned ldsoff = (unsigned)(uintptr_t)&tile[ty + cc * 8][tx];
        asm volatile("global_load_async_to_lds_b32 %0, %1, off"
                     :: "v"(ldsoff), "v"(gp) : "memory");
    }
    asm volatile("s_wait_asynccnt 0x0" ::: "memory");
    __syncthreads();
#pragma unroll
    for (int cc = 0; cc < 4; ++cc) {
        int row = ty + cc * 8;
        A0[((size_t)b * HWPIX + hw0 + row) * C_DIM + c0 + tx] = f2bf(tile[tx][row]);
    }
}

// ---------------------------------------------------------------------------
// 2) f32 -> bf16 weight conversion
// ---------------------------------------------------------------------------
__global__ __launch_bounds__(256)
void cvt_bf16_kernel(const float* __restrict__ in, unsigned short* __restrict__ out, int n) {
    int i = blockIdx.x * 256 + threadIdx.x;
    if (i < n) out[i] = f2bf(in[i]);
}

// ---------------------------------------------------------------------------
// 3) WMMA bf16 GEMM: out[M,N] = A[M,K] * B[N,K]^T  (+epilogue)
//    Wave tile 32x64 (2x4 fragments, 8 WMMA / K-step); block tile 128x128.
//    EPI 0: f32 out + bias            (cv1)
//    EPI 1: f32 out                   (in_proj, x_proj)
//    EPI 2: bf16 out                  (out_proj)
//    EPI 3: channel-major f32 + bias + residual x   (cv2, final output)
//    PARTIAL: scalarized (readfirstlane -> s_cbranch) fragment masking, only
//    used for x_proj (N=48); full GEMMs get a branch-free inner loop.
// ---------------------------------------------------------------------------
template<int EPI, bool PARTIAL>
__global__ __launch_bounds__(256)
void gemm_wmma(const unsigned short* __restrict__ A, const unsigned short* __restrict__ B,
               const float* __restrict__ bias, void* __restrict__ out,
               int M, int N, int K, const float* __restrict__ resid) {
    int lane = threadIdx.x & 31;
    int wave = threadIdx.x >> 5;
    int wm = wave >> 1, wn = wave & 1;
    int m0 = blockIdx.y * 128 + wm * 32;
    int n0 = blockIdx.x * 128 + wn * 64;
    int half = lane >> 4;
    int r    = lane & 15;

    int bv[4];
#pragma unroll
    for (int j = 0; j < 4; ++j)
        bv[j] = PARTIAL ? __builtin_amdgcn_readfirstlane((n0 + j * 16 + 16) <= N ? 1 : 0) : 1;

    v8f acc[2][4];
#pragma unroll
    for (int i = 0; i < 2; ++i)
#pragma unroll
        for (int j = 0; j < 4; ++j)
            acc[i][j] = (v8f){0, 0, 0, 0, 0, 0, 0, 0};

    // A fragment: lane holds row m0+r, 16 contiguous K at k + half*16
    const unsigned short* Abase[2];
    Abase[0] = A + (size_t)(m0 + r) * K + half * 16;
    Abase[1] = Abase[0] + (size_t)16 * K;
    // B fragment: lane holds row n0+j*16+r (= N index), 16 contiguous K at k + half*16
    const unsigned short* Bbase[4];
#pragma unroll
    for (int j = 0; j < 4; ++j)
        Bbase[j] = B + (size_t)(n0 + j * 16 + r) * K + half * 16;

    for (int k = 0; k < K; k += 32) {
        FragU a[2], bf[4];
#pragma unroll
        for (int i = 0; i < 2; ++i) {
            const uint4* p = (const uint4*)(Abase[i] + k);
            a[i].u[0] = p[0]; a[i].u[1] = p[1];
        }
#pragma unroll
        for (int j = 0; j < 4; ++j) {
            if (PARTIAL && !bv[j]) continue;
            const uint4* p = (const uint4*)(Bbase[j] + k);
            bf[j].u[0] = p[0]; bf[j].u[1] = p[1];
        }
        if (k + 64 < K) {
            __builtin_prefetch(Abase[0] + k + 64, 0, 1);
            __builtin_prefetch(Abase[1] + k + 64, 0, 1);
        }
#pragma unroll
        for (int j = 0; j < 4; ++j) {
            if (PARTIAL && !bv[j]) continue;
            acc[0][j] = __builtin_amdgcn_wmma_f32_16x16x32_bf16(false, a[0].v, false, bf[j].v,
                                                                (short)0, acc[0][j], false, false);
            acc[1][j] = __builtin_amdgcn_wmma_f32_16x16x32_bf16(false, a[1].v, false, bf[j].v,
                                                                (short)0, acc[1][j], false, false);
        }
    }

#pragma unroll
    for (int j = 0; j < 4; ++j) {
        if (PARTIAL && !bv[j]) continue;
        int col = n0 + j * 16 + r;
#pragma unroll
        for (int i = 0; i < 2; ++i) {
            int mbase = m0 + i * 16 + half * 8;
#pragma unroll
            for (int v = 0; v < 8; ++v) {
                float val = acc[i][j][v];
                int row = mbase + v;
                if (EPI == 0) {
                    ((float*)out)[(size_t)row * N + col] = val + bias[col];
                } else if (EPI == 1) {
                    ((float*)out)[(size_t)row * N + col] = val;
                } else if (EPI == 2) {
                    ((unsigned short*)out)[(size_t)row * N + col] = f2bf(val);
                } else {
                    int bb = row >> 12, hw = row & (HWPIX - 1);
                    size_t addr = ((size_t)(bb * C_DIM + col)) * HWPIX + hw;
                    ((float*)out)[addr] = val + bias[col] + resid[addr];
                }
            }
        }
    }
}

// ---------------------------------------------------------------------------
// 4) LayerNorm over C=256 per row; one wave per row; bf16 output
// ---------------------------------------------------------------------------
__global__ __launch_bounds__(256)
void layernorm_kernel(const float* __restrict__ t0, const float* __restrict__ g,
                      const float* __restrict__ bta, unsigned short* __restrict__ out) {
    int wave = threadIdx.x >> 5, lane = threadIdx.x & 31;
    int row = blockIdx.x * 8 + wave;
    const float* p = t0 + (size_t)row * C_DIM;
    float v[8], s = 0.f, s2 = 0.f;
#pragma unroll
    for (int i = 0; i < 8; ++i) { v[i] = p[lane + i * 32]; s += v[i]; s2 += v[i] * v[i]; }
#pragma unroll
    for (int off = 16; off > 0; off >>= 1) {
        s  += __shfl_xor(s,  off, 32);
        s2 += __shfl_xor(s2, off, 32);
    }
    float mean = s * (1.f / C_DIM);
    float var  = s2 * (1.f / C_DIM) - mean * mean;
    float rstd = rsqrtf(var + 1e-5f);
    unsigned short* o = out + (size_t)row * C_DIM;
#pragma unroll
    for (int i = 0; i < 8; ++i) {
        int c = lane + i * 32;
        o[c] = f2bf((v[i] - mean) * rstd * g[c] + bta[c]);
    }
}

// ---------------------------------------------------------------------------
// 5) depthwise causal conv (D_CONV=4) + SiLU ; outputs u in f32 and bf16
// ---------------------------------------------------------------------------
__global__ __launch_bounds__(256)
void conv_silu_kernel(const float* __restrict__ xz, const float* __restrict__ cw,
                      const float* __restrict__ cb, float* __restrict__ u32,
                      unsigned short* __restrict__ u16) {
    int idx = blockIdx.x * 256 + threadIdx.x;   // NPIX * DINNER
    int d = idx & (DINNER - 1);
    int n = idx >> 9;
    int l = n & (HWPIX - 1);
    float acc = cb[d];
#pragma unroll
    for (int k = 0; k < 4; ++k) {
        int ls = l - 3 + k;
        if (ls >= 0) acc += cw[d * 4 + k] * xz[(size_t)(n - 3 + k) * (2 * DINNER) + d];
    }
    float s = acc / (1.f + __expf(-acc));
    u32[idx] = s;
    u16[idx] = f2bf(s);
}

// ---------------------------------------------------------------------------
// 6) dt = softplus(x_dbl[:, :16] @ dt_proj_w^T + dt_proj_b)   (K=16, VALU)
// ---------------------------------------------------------------------------
__global__ __launch_bounds__(256)
void dt_kernel(const float* __restrict__ xdbl, const float* __restrict__ w,
               const float* __restrict__ bias, float* __restrict__ dtv) {
    int idx = blockIdx.x * 256 + threadIdx.x;
    int d = idx & (DINNER - 1);
    int n = idx >> 9;
    const float* row = xdbl + (size_t)n * XD;
    float acc = bias[d];
#pragma unroll
    for (int r = 0; r < DTRANK; ++r) acc += row[r] * w[d * DTRANK + r];
    dtv[idx] = (acc > 20.f) ? acc : log1pf(__expf(acc));
}

// ---------------------------------------------------------------------------
// 7) selective scan: 16 lanes per channel (one per state), shfl_xor reduce.
//    Software-pipelined: step l+1 loads issue before step l's reduce chain.
//    Fused epilogue: y = scan + u*D, y *= silu(z). Output bf16 for out_proj.
// ---------------------------------------------------------------------------
__global__ __launch_bounds__(256)
void scan_kernel(const float* __restrict__ dtv, const float* __restrict__ u32,
                 const float* __restrict__ xdbl, const float* __restrict__ xz,
                 const float* __restrict__ A_log, const float* __restrict__ Dv,
                 unsigned short* __restrict__ y16) {
    int b = blockIdx.y;
    int d = blockIdx.x * 16 + (threadIdx.x >> 4);
    int n = threadIdx.x & 15;
    float A  = -__expf(A_log[d * DSTATE + n]);
    float Dd = Dv[d];
    float h = 0.f;
    size_t rb = (size_t)b * HWPIX;

    float dt = dtv[rb * DINNER + d];
    float u  = u32[rb * DINNER + d];
    float Bt = xdbl[rb * XD + DTRANK + n];
    float Ct = xdbl[rb * XD + DTRANK + DSTATE + n];

    for (int l = 0; l < HWPIX; ++l) {
        size_t row = rb + l;
        float dtn = 0.f, un = 0.f, Btn = 0.f, Ctn = 0.f;
        if (l + 1 < HWPIX) {
            size_t nr = row + 1;
            dtn = dtv[nr * DINNER + d];
            un  = u32[nr * DINNER + d];
            Btn = xdbl[nr * XD + DTRANK + n];
            Ctn = xdbl[nr * XD + DTRANK + DSTATE + n];
        }
        float dA = __expf(dt * A);
        h = dA * h + dt * Bt * u;
        float c = h * Ct;
        c += __shfl_xor(c, 8, 16);
        c += __shfl_xor(c, 4, 16);
        c += __shfl_xor(c, 2, 16);
        c += __shfl_xor(c, 1, 16);
        if (n == 0) {
            float y = c + u * Dd;
            float z = xz[row * (2 * DINNER) + DINNER + d];
            y *= z / (1.f + __expf(-z));
            y16[row * DINNER + d] = f2bf(y);
        }
        dt = dtn; u = un; Bt = Btn; Ct = Ctn;
    }
}

// ---------------------------------------------------------------------------
extern "C" void kernel_launch(void* const* d_in, const int* in_sizes, int n_in,
                              void* d_out, int out_size, void* d_ws, size_t ws_size,
                              hipStream_t stream) {
    const float* x        = (const float*)d_in[0];
    const float* cv1_w    = (const float*)d_in[1];
    const float* cv1_b    = (const float*)d_in[2];
    const float* ln_g     = (const float*)d_in[3];
    const float* ln_b     = (const float*)d_in[4];
    const float* inp_w    = (const float*)d_in[5];
    const float* conv_w   = (const float*)d_in[6];
    const float* conv_b   = (const float*)d_in[7];
    const float* xp_w     = (const float*)d_in[8];
    const float* dtp_w    = (const float*)d_in[9];
    const float* dtp_b    = (const float*)d_in[10];
    const float* A_log    = (const float*)d_in[11];
    const float* Dv       = (const float*)d_in[12];
    const float* outp_w   = (const float*)d_in[13];
    const float* cv2_w    = (const float*)d_in[14];
    const float* cv2_b    = (const float*)d_in[15];
    float* out = (float*)d_out;

    char* ws = (char*)d_ws;
    size_t off = 0;
    auto alloc = [&](size_t bytes) { size_t o = off; off += (bytes + 255) & ~(size_t)255; return o; };

    unsigned short* A0    = (unsigned short*)(ws + alloc((size_t)NPIX * C_DIM * 2));
    float*          t0    = (float*)         (ws + alloc((size_t)NPIX * C_DIM * 4));
    unsigned short* A1    = (unsigned short*)(ws + alloc((size_t)NPIX * C_DIM * 2));
    float*          xzb   = (float*)         (ws + alloc((size_t)NPIX * 2 * DINNER * 4));
    float*          u32   = (float*)         (ws + alloc((size_t)NPIX * DINNER * 4));
    unsigned short* u16   = (unsigned short*)(ws + alloc((size_t)NPIX * DINNER * 2));
    float*          xdbl  = (float*)         (ws + alloc((size_t)NPIX * XD * 4));
    float*          dtv   = (float*)         (ws + alloc((size_t)NPIX * DINNER * 4));
    unsigned short* y16   = (unsigned short*)(ws + alloc((size_t)NPIX * DINNER * 2));
    unsigned short* t1    = (unsigned short*)(ws + alloc((size_t)NPIX * C_DIM * 2));
    unsigned short* wcv1  = (unsigned short*)(ws + alloc((size_t)C_DIM * C_DIM * 2));
    unsigned short* winp  = (unsigned short*)(ws + alloc((size_t)2 * DINNER * C_DIM * 2));
    unsigned short* wxp   = (unsigned short*)(ws + alloc((size_t)XD * DINNER * 2));
    unsigned short* wop   = (unsigned short*)(ws + alloc((size_t)C_DIM * DINNER * 2));
    unsigned short* wcv2  = (unsigned short*)(ws + alloc((size_t)C_DIM * C_DIM * 2));

    // weight conversions f32 -> bf16
    cvt_bf16_kernel<<<(C_DIM * C_DIM + 255) / 256, 256, 0, stream>>>(cv1_w,  wcv1, C_DIM * C_DIM);
    cvt_bf16_kernel<<<(2 * DINNER * C_DIM + 255) / 256, 256, 0, stream>>>(inp_w, winp, 2 * DINNER * C_DIM);
    cvt_bf16_kernel<<<(XD * DINNER + 255) / 256, 256, 0, stream>>>(xp_w,   wxp,  XD * DINNER);
    cvt_bf16_kernel<<<(C_DIM * DINNER + 255) / 256, 256, 0, stream>>>(outp_w, wop, C_DIM * DINNER);
    cvt_bf16_kernel<<<(C_DIM * C_DIM + 255) / 256, 256, 0, stream>>>(cv2_w,  wcv2, C_DIM * C_DIM);

    // x -> pixel-major bf16 (async global->LDS staging)
    transpose_bf16_kernel<<<dim3(HWPIX / 32, C_DIM / 32, BATCH), dim3(32, 8), 0, stream>>>(x, A0);

    // cv1: t0 = A0 @ cv1_w^T + b
    gemm_wmma<0, false><<<dim3(C_DIM / 128, NPIX / 128), 256, 0, stream>>>(
        A0, wcv1, cv1_b, t0, NPIX, C_DIM, C_DIM, nullptr);
    // layernorm -> bf16
    layernorm_kernel<<<NPIX / 8, 256, 0, stream>>>(t0, ln_g, ln_b, A1);

    // in_proj: xz = A1 @ in_proj_w^T  (N=1024)
    gemm_wmma<1, false><<<dim3(2 * DINNER / 128, NPIX / 128), 256, 0, stream>>>(
        A1, winp, nullptr, xzb, NPIX, 2 * DINNER, C_DIM, nullptr);
    // depthwise conv + silu
    conv_silu_kernel<<<(NPIX * DINNER) / 256, 256, 0, stream>>>(xzb, conv_w, conv_b, u32, u16);

    // x_proj: xdbl = u @ x_proj_w^T  (N=48, scalar-masked partial tiles)
    gemm_wmma<1, true><<<dim3(1, NPIX / 128), 256, 0, stream>>>(
        u16, wxp, nullptr, xdbl, NPIX, XD, DINNER, nullptr);
    // dt path (K=16, VALU)
    dt_kernel<<<(NPIX * DINNER) / 256, 256, 0, stream>>>(xdbl, dtp_w, dtp_b, dtv);

    // selective scan + gating -> y bf16
    scan_kernel<<<dim3(DINNER / 16, BATCH), 256, 0, stream>>>(dtv, u32, xdbl, xzb, A_log, Dv, y16);

    // out_proj: t1 = y @ out_proj_w^T  -> bf16
    gemm_wmma<2, false><<<dim3(C_DIM / 128, NPIX / 128), 256, 0, stream>>>(
        y16, wop, nullptr, t1, NPIX, C_DIM, DINNER, nullptr);
    // cv2 + bias + residual, scattered back to (b, c, h, w)
    gemm_wmma<3, false><<<dim3(C_DIM / 128, NPIX / 128), 256, 0, stream>>>(
        t1, wcv2, cv2_b, out, NPIX, C_DIM, C_DIM, x);
    (void)in_sizes; (void)n_in; (void)out_size; (void)ws_size;
}
